// MultiHeadAttention_59072980189725
// MI455X (gfx1250) — matrix operand order repaired and checked by reading, not compiled, not run on hardware
//
#include <hip/hip_runtime.h>

typedef __attribute__((ext_vector_type(16))) _Float16 v16h;
typedef __attribute__((ext_vector_type(8)))  _Float16 v8h;
typedef __attribute__((ext_vector_type(8)))  float    v8f;

#define BB 4
#define LL 2048
#define DD 512
#define HH 8
#define DKK 64

#define USE_ASYNC_LDS 1

// ---- WMMA fragment helpers (CDNA5 16x16x32 f16, wave32) ----
// A-matrix 16x32 (MxK): lanes 0-15: h0..7=K0..7, h8..15=K16..23 ; lanes 16-31: +8
__device__ __forceinline__ v16h load_a_frag(const _Float16* row_base, int hi) {
  v8h lo = *(const v8h*)(row_base + hi * 8);
  v8h hh = *(const v8h*)(row_base + 16 + hi * 8);
  v16h r;
#pragma unroll
  for (int i = 0; i < 8; ++i) { r[i] = lo[i]; r[i + 8] = hh[i]; }
  return r;
}
// B-matrix 32x16 (KxN): lane holds col n=lane%16; same K order as A:
// lanes 0-15 -> K0..7,K16..23, lanes 16-31 -> K8..15,K24..31.
__device__ __forceinline__ v16h load_b_frag(const _Float16* col_base, int hi) {
  v8h a0 = *(const v8h*)(col_base + hi * 8);
  v8h a1 = *(const v8h*)(col_base + 16 + hi * 8);
  v16h r;
#pragma unroll
  for (int i = 0; i < 8; ++i) { r[i] = a0[i]; r[i + 8] = a1[i]; }
  return r;
}
__device__ __forceinline__ v8f wmma_f16(v16h a, v16h b, v8f c) {
  return __builtin_amdgcn_wmma_f32_16x16x32_f16(false, a, false, b, (short)0, c,
                                                false, false);
}

#if USE_ASYNC_LDS
// GLOBAL_LOAD_ASYNC_TO_LDS_B128 (GVS mode): LDS[vdst] = MEM[saddr + voff].
// LDS byte address = low 32 bits of the generic pointer (aperture truncation).
__device__ __forceinline__ void async_ld_b128(unsigned long long sbase,
                                              unsigned voff, const void* lds) {
  unsigned l = (unsigned)(unsigned long long)lds;
  asm volatile("global_load_async_to_lds_b128 %0, %1, %2"
               :: "v"(l), "v"(voff), "s"(sbase) : "memory");
}
__device__ __forceinline__ void wait_async0() {
  asm volatile("s_wait_asynccnt 0x0" ::: "memory");
}
#endif

// ---- small staging helpers for the GEMMs ----
__device__ __forceinline__ void ld16f(float4* r, const float* src) {
#pragma unroll
  for (int u = 0; u < 4; ++u) r[u] = ((const float4*)src)[u];
}
__device__ __forceinline__ void cvt_store16(_Float16* d, const float4* r) {
#pragma unroll
  for (int u = 0; u < 4; ++u) {
    float4 f = r[u];
    d[4 * u + 0] = (_Float16)f.x; d[4 * u + 1] = (_Float16)f.y;
    d[4 * u + 2] = (_Float16)f.z; d[4 * u + 3] = (_Float16)f.w;
  }
}
__device__ __forceinline__ void mm_step(const _Float16* A, const _Float16* B,
                                        int mw, int nw, int ln, int hi,
                                        v8f acc[2][2]) {
  v16h af[2], bf[2];
#pragma unroll
  for (int i = 0; i < 2; ++i)
    af[i] = load_a_frag(A + (32 * mw + 16 * i + ln) * 40, hi);
#pragma unroll
  for (int j = 0; j < 2; ++j)
    bf[j] = load_b_frag(B + (32 * nw + 16 * j + ln) * 40, hi);
#pragma unroll
  for (int i = 0; i < 2; ++i)
#pragma unroll
    for (int j = 0; j < 2; ++j) acc[i][j] = wmma_f16(af[i], bf[j], acc[i][j]);
}

// ============================================================================
// Kernel 1: QKV projection  y = x @ W^T + b  (fp32 in -> f16 per-head out)
// grid: (8192/64, 8, 3)  block: 128.  64x64 tile, K-step 32, LDS ping-pong.
// z=0 -> Qh[B,H,L,DK], z=1 -> Kh[B,H,L,DK], z=2 -> Vt[B,H,DK,L]
// ============================================================================
__global__ __launch_bounds__(128) void gemm_qkv_kernel(
    const float* __restrict__ xq, const float* __restrict__ xk,
    const float* __restrict__ xv, const float* __restrict__ wq,
    const float* __restrict__ wk, const float* __restrict__ wv,
    const float* __restrict__ bq, const float* __restrict__ bk,
    const float* __restrict__ bv, _Float16* __restrict__ Qh,
    _Float16* __restrict__ Kh, _Float16* __restrict__ Vt) {
  const int z = blockIdx.z;
  const float* x    = (z == 0) ? xq : (z == 1) ? xk : xv;
  const float* w    = (z == 0) ? wq : (z == 1) ? wk : wv;
  const float* bias = (z == 0) ? bq : (z == 1) ? bk : bv;

  __shared__ _Float16 Ah[2][64 * 40];
  __shared__ _Float16 Bh[2][64 * 40];

  const int tid = threadIdx.x;
  const int lane = tid & 31, wave = tid >> 5;
  const int mw = wave >> 1, nw = wave & 1;
  const int hi = (lane >> 4) & 1, ln = lane & 15;
  const int m0 = blockIdx.x * 64;
  const int n0 = blockIdx.y * 64;
  const int lrow = tid >> 1;
  const int lseg = (tid & 1) * 16;

  const float* xrow = x + (m0 + lrow) * DD + lseg;
  const float* wrow = w + (n0 + lrow) * DD + lseg;
  _Float16* da0 = Ah[0] + lrow * 40 + lseg;
  _Float16* db0 = Bh[0] + lrow * 40 + lseg;
  _Float16* da1 = Ah[1] + lrow * 40 + lseg;
  _Float16* db1 = Bh[1] + lrow * 40 + lseg;

  v8f acc[2][2] = {};
  float4 ra[4], rb[4], na[4], nb[4];
  ld16f(ra, xrow); ld16f(rb, wrow);

  for (int kb = 0; kb < DD; kb += 64) {
    // stage 0: tile kb (regs ra/rb) -> buffer 0
    cvt_store16(da0, ra); cvt_store16(db0, rb);
    ld16f(na, xrow + kb + 32); ld16f(nb, wrow + kb + 32);  // prefetch kb+32
    __syncthreads();
    mm_step(Ah[0], Bh[0], mw, nw, ln, hi, acc);
    // stage 1: tile kb+32 (regs na/nb) -> buffer 1
    cvt_store16(da1, na); cvt_store16(db1, nb);
    if (kb + 64 < DD) { ld16f(ra, xrow + kb + 64); ld16f(rb, wrow + kb + 64); }
    __syncthreads();
    mm_step(Ah[1], Bh[1], mw, nw, ln, hi, acc);
  }

  // branchless epilogue: one base + uniform strides per z
  const int h = blockIdx.y;
  const int bimg = m0 >> 11;            // tile never crosses batch boundary
  const int pos0 = m0 & (LL - 1);
  _Float16* dst = (z == 0) ? Qh : (z == 1) ? Kh : Vt;
  const int posStride = (z == 2) ? 1 : DKK;
  const int colStride = (z == 2) ? LL : 1;
  const int base = (bimg * HH + h) * (LL * DKK);
#pragma unroll
  for (int i = 0; i < 2; ++i)
#pragma unroll
    for (int j = 0; j < 2; ++j) {
      const int colL = 32 * nw + 16 * j + ln;
      const float bvv = bias[n0 + colL];
      const int cterm = base + colL * colStride;
#pragma unroll
      for (int r = 0; r < 8; ++r) {
        const int pos = pos0 + 32 * mw + 16 * i + r + hi * 8;
        dst[cterm + pos * posStride] = (_Float16)(acc[i][j][r] + bvv);
      }
    }
}

// ============================================================================
// Kernel 2: flash attention, grid (L/64, H, B), block 128 (4 waves x 16 q).
// K/V chunks double-buffered in LDS via async global->LDS copies (ASYNCcnt).
// ============================================================================
__global__ __launch_bounds__(128) void attn_kernel(
    const _Float16* __restrict__ Qh, const _Float16* __restrict__ Kh,
    const _Float16* __restrict__ Vt, const int* __restrict__ masks,
    _Float16* __restrict__ Ctx) {
  __shared__ _Float16 Ks[2][64 * 72];      // [key][dk]
  __shared__ _Float16 Vs[2][64 * 72];      // [dk][key]
  __shared__ _Float16 Ps[4 * 16 * 72];     // per-wave P [q][key]
  __shared__ float mbv[2][64];

  const int tid = threadIdx.x, lane = tid & 31, wave = tid >> 5;
  const int hi = (lane >> 4) & 1, ln = lane & 15;
  const int b = blockIdx.z, h = blockIdx.y;
  const int bh = b * HH + h;
  const int q0 = blockIdx.x * 64 + wave * 16;
  const int lrow = tid >> 1;
  const int lseg = (tid & 1) * 32;

  const _Float16* Kb = Kh + bh * (LL * DKK);
  const _Float16* Vb = Vt + bh * (DKK * LL);
#if USE_ASYNC_LDS
  const unsigned long long Kbase = (unsigned long long)Kb;
  const unsigned long long Vbase = (unsigned long long)Vb;
#endif

  v16h qfrag[2];
  {
    const _Float16* qrow = Qh + (bh * LL + q0 + ln) * DKK;
    qfrag[0] = load_a_frag(qrow, hi);
    qfrag[1] = load_a_frag(qrow + 32, hi);
  }

  float m[8], lsum[8];
  v8f acc[4] = {};
#pragma unroll
  for (int r = 0; r < 8; ++r) { m[r] = -3.0e38f; lsum[r] = 0.0f; }
  _Float16* Pw = Ps + wave * 16 * 72;

  auto issue_chunk = [&](int kc, int buf) {
#if USE_ASYNC_LDS
    const unsigned gk = (unsigned)(((kc + lrow) * DKK + lseg) * 2);
    const unsigned gv = (unsigned)((lrow * LL + kc + lseg) * 2);
    const _Float16* lk = Ks[buf] + lrow * 72 + lseg;
    const _Float16* lv = Vs[buf] + lrow * 72 + lseg;
#pragma unroll
    for (int u = 0; u < 4; ++u) {
      async_ld_b128(Kbase, gk + 16u * u, lk + 8 * u);
      async_ld_b128(Vbase, gv + 16u * u, lv + 8 * u);
    }
#else
    const v8h* sk = (const v8h*)(Kb + (kc + lrow) * DKK + lseg);
    const v8h* sv = (const v8h*)(Vb + lrow * LL + kc + lseg);
    _Float16* lk = Ks[buf] + lrow * 72 + lseg;
    _Float16* lv = Vs[buf] + lrow * 72 + lseg;
#pragma unroll
    for (int u = 0; u < 4; ++u) {
      *(v8h*)(lk + 8 * u) = sk[u];
      *(v8h*)(lv + 8 * u) = sv[u];
    }
#endif
  };

  // prologue: chunk 0 in flight, mask bias for chunk 0
  issue_chunk(0, 0);
  if (tid < 64) mbv[0][tid] = masks[b * LL + tid] ? 0.0f : -3.0e38f;

  int p = 0;
  for (int kc = 0; kc < LL; kc += 64) {
#if USE_ASYNC_LDS
    wait_async0();
#endif
    __syncthreads();  // buffer p (+ mbv[p]) now valid for all waves
    const int pn = p ^ 1;
    if (kc + 64 < LL) {
      issue_chunk(kc + 64, pn);
      if (tid < 64)
        mbv[pn][tid] = masks[b * LL + kc + 64 + tid] ? 0.0f : -3.0e38f;
    }

    const _Float16* Kp = Ks[p];
    const _Float16* Vp = Vs[p];

    // scores S = (Q K^T) * scale + mask_bias
    v8f S[4];
#pragma unroll
    for (int t = 0; t < 4; ++t) {
      v8f s = {};
      const _Float16* kbase = Kp + (16 * t + ln) * 72;
      s = wmma_f16(qfrag[0], load_b_frag(kbase, hi), s);
      s = wmma_f16(qfrag[1], load_b_frag(kbase + 32, hi), s);
      const float bias = mbv[p][16 * t + ln];
#pragma unroll
      for (int r = 0; r < 8; ++r) S[t][r] = s[r] * 0.125f + bias;
    }
    // per-row chunk max (reduce across the 16-lane column group)
    float cmax[8];
#pragma unroll
    for (int r = 0; r < 8; ++r)
      cmax[r] = fmaxf(fmaxf(S[0][r], S[1][r]), fmaxf(S[2][r], S[3][r]));
#pragma unroll
    for (int off = 1; off < 16; off <<= 1)
#pragma unroll
      for (int r = 0; r < 8; ++r)
        cmax[r] = fmaxf(cmax[r], __shfl_xor(cmax[r], off, 32));
    // online rescale
    float corr[8];
#pragma unroll
    for (int r = 0; r < 8; ++r) {
      float mn = fmaxf(m[r], cmax[r]);
      corr[r] = __expf(m[r] - mn);
      m[r] = mn;
      lsum[r] *= corr[r];
    }
#pragma unroll
    for (int d = 0; d < 4; ++d)
#pragma unroll
      for (int r = 0; r < 8; ++r) acc[d][r] *= corr[r];
    // P = exp(S - m); per-wave LDS roundtrip for C->A layout transpose
    float pl[8] = {0, 0, 0, 0, 0, 0, 0, 0};
#pragma unroll
    for (int t = 0; t < 4; ++t)
#pragma unroll
      for (int r = 0; r < 8; ++r) {
        float pv = __expf(S[t][r] - m[r]);
        pl[r] += pv;
        Pw[(r + hi * 8) * 72 + 16 * t + ln] = (_Float16)pv;
      }
#pragma unroll
    for (int off = 1; off < 16; off <<= 1)
#pragma unroll
      for (int r = 0; r < 8; ++r) pl[r] += __shfl_xor(pl[r], off, 32);
#pragma unroll
    for (int r = 0; r < 8; ++r) lsum[r] += pl[r];

    // ctx += P @ V  (same-wave DS ops are in-order: store->load is safe)
    v16h pf0 = load_a_frag(Pw + ln * 72, hi);
    v16h pf1 = load_a_frag(Pw + ln * 72 + 32, hi);
#pragma unroll
    for (int d = 0; d < 4; ++d) {
      const _Float16* vbase = Vp + (16 * d + ln) * 72;
      acc[d] = wmma_f16(pf0, load_b_frag(vbase, hi), acc[d]);
      acc[d] = wmma_f16(pf1, load_b_frag(vbase + 32, hi), acc[d]);
    }
    p = pn;
  }

  float rl[8];
#pragma unroll
  for (int r = 0; r < 8; ++r) rl[r] = 1.0f / lsum[r];
#pragma unroll
  for (int d = 0; d < 4; ++d)
#pragma unroll
    for (int r = 0; r < 8; ++r) {
      const int qi = q0 + r + hi * 8;
      const int col = h * DKK + 16 * d + ln;
      Ctx[(b * LL + qi) * DD + col] = (_Float16)(acc[d][r] * rl[r]);
    }
}

// ============================================================================
// Kernel 3: output projection  out = ctx @ wo^T + wo_b  (f16 in -> fp32 out)
// grid: (8192/64, 512/64)  block: 128.  LDS ping-pong like kernel 1.
// ============================================================================
__global__ __launch_bounds__(128) void gemm_out_kernel(
    const _Float16* __restrict__ ctx, const float* __restrict__ w,
    const float* __restrict__ bias, float* __restrict__ out) {
  __shared__ _Float16 Ah[2][64 * 40];
  __shared__ _Float16 Bh[2][64 * 40];

  const int tid = threadIdx.x;
  const int lane = tid & 31, wave = tid >> 5;
  const int mw = wave >> 1, nw = wave & 1;
  const int hi = (lane >> 4) & 1, ln = lane & 15;
  const int m0 = blockIdx.x * 64;
  const int n0 = blockIdx.y * 64;
  const int lrow = tid >> 1;
  const int lseg = (tid & 1) * 16;

  const _Float16* arow = ctx + (m0 + lrow) * DD + lseg;
  const float* wrow = w + (n0 + lrow) * DD + lseg;
  _Float16* da0 = Ah[0] + lrow * 40 + lseg;
  _Float16* db0 = Bh[0] + lrow * 40 + lseg;
  _Float16* da1 = Ah[1] + lrow * 40 + lseg;
  _Float16* db1 = Bh[1] + lrow * 40 + lseg;

  v8f acc[2][2] = {};
  v8h ca[2], na[2];
  float4 rb[4], nb[4];
  ca[0] = ((const v8h*)arow)[0]; ca[1] = ((const v8h*)arow)[1];
  ld16f(rb, wrow);

  for (int kb = 0; kb < DD; kb += 64) {
    *(v8h*)(da0) = ca[0]; *(v8h*)(da0 + 8) = ca[1];
    cvt_store16(db0, rb);
    na[0] = ((const v8h*)(arow + kb + 32))[0];
    na[1] = ((const v8h*)(arow + kb + 32))[1];
    ld16f(nb, wrow + kb + 32);
    __syncthreads();
    mm_step(Ah[0], Bh[0], mw, nw, ln, hi, acc);

    *(v8h*)(da1) = na[0]; *(v8h*)(da1 + 8) = na[1];
    cvt_store16(db1, nb);
    if (kb + 64 < DD) {
      ca[0] = ((const v8h*)(arow + kb + 64))[0];
      ca[1] = ((const v8h*)(arow + kb + 64))[1];
      ld16f(rb, wrow + kb + 64);
    }
    __syncthreads();
    mm_step(Ah[1], Bh[1], mw, nw, ln, hi, acc);
  }

#pragma unroll
  for (int i = 0; i < 2; ++i)
#pragma unroll
    for (int j = 0; j < 2; ++j) {
      const int col = n0 + 32 * nw + 16 * j + ln;
      const float bvv = bias[col];
#pragma unroll
      for (int r = 0; r < 8; ++r) {
        const int token = m0 + 32 * mw + 16 * i + r + hi * 8;
        out[token * DD + col] = acc[i][j][r] + bvv;
      }
    }
}

extern "C" void kernel_launch(void* const* d_in, const int* in_sizes, int n_in,
                              void* d_out, int out_size, void* d_ws,
                              size_t ws_size, hipStream_t stream) {
  (void)in_sizes; (void)n_in; (void)out_size; (void)ws_size;
  const float* q = (const float*)d_in[0];
  const float* k = (const float*)d_in[1];
  const float* v = (const float*)d_in[2];
  const int* masks = (const int*)d_in[3];
  const float* wq_w = (const float*)d_in[4];
  const float* wq_b = (const float*)d_in[5];
  const float* wk_w = (const float*)d_in[6];
  const float* wk_b = (const float*)d_in[7];
  const float* wv_w = (const float*)d_in[8];
  const float* wv_b = (const float*)d_in[9];
  const float* wo_w = (const float*)d_in[10];
  const float* wo_b = (const float*)d_in[11];
  float* out = (float*)d_out;

  const size_t nper = (size_t)BB * HH * LL * DKK;  // 8 MB f16 each
  _Float16* Qh = (_Float16*)d_ws;
  _Float16* Kh = Qh + nper;
  _Float16* Vt = Kh + nper;
  _Float16* Cx = Vt + nper;  // total ws use = 32 MB

  gemm_qkv_kernel<<<dim3((BB * LL) / 64, DD / 64, 3), 128, 0, stream>>>(
      q, k, v, wq_w, wk_w, wv_w, wq_b, wk_b, wv_b, Qh, Kh, Vt);
  attn_kernel<<<dim3(LL / 64, HH, BB), 128, 0, stream>>>(Qh, Kh, Vt, masks, Cx);
  gemm_out_kernel<<<dim3((BB * LL) / 64, DD / 64), 128, 0, stream>>>(Cx, wo_w,
                                                                     wo_b, out);
}
